// MatchGraphEncoder_28338194219475
// MI455X (gfx1250) — matrix operand order repaired and checked by reading, not compiled, not run on hardware
//
#include <hip/hip_runtime.h>
#include <hip/hip_bf16.h>
#include <math.h>

// ---------------------------------------------------------------------------
// 3-layer GAT encoder for MI455X (gfx1250, wave32).
// GEMMs use v_wmma_f32_16x16x32_bf16 (bf16 in, f32 accum).
// Edge softmax: atomic-max -> exp/atomic-add denom -> fused normalize+scatter.
// h stored bf16 to halve the dominant edge-gather traffic.
// b1/b2 omitted: per-channel constants cancel in BatchNorm mean subtraction.
// ---------------------------------------------------------------------------

#define Nn    50000
#define Ee    400000
#define ETOTe 450000   // E + N self loops

typedef __attribute__((ext_vector_type(16))) __bf16 v16bf;
typedef __attribute__((ext_vector_type(8)))  __bf16 v8bf;
typedef __attribute__((ext_vector_type(8)))  float  v8f;

// ---- workspace layout (bytes, all 256-aligned) ----------------------------
#define OFF_BFIN  ((size_t)0)                       // N*128 bf16 = 12,800,000
#define OFF_BFW   (OFF_BFIN + 12800000)             // 512*128 bf16 = 131,072
#define OFF_H     (OFF_BFW  + 131072)               // N*512 bf16 = 51,200,000
#define OFF_ASRC  (OFF_H    + 51200000)             // N*4 f32 = 800,000
#define OFF_ADST  (OFF_ASRC + 800000)
#define OFF_M     (OFF_ADST + 800000)
#define OFF_DEN   (OFF_M    + 800000)
#define OFF_E     (OFF_DEN  + 800000)               // ETOT*4 f32 = 7,200,000
#define OFF_Y     (OFF_E    + 7200000)              // N*128 f32 = 25,600,000
#define OFF_ST    (OFF_Y    + 25600000)             // 256 f32

// ---------------------------------------------------------------------------

__device__ __forceinline__ float atomicMaxFloat(float* addr, float v) {
  // monotonic int/uint mapping; valid with -inf init and mixed signs
  if (v >= 0.0f)
    return __int_as_float(atomicMax((int*)addr, __float_as_int(v)));
  else
    return __uint_as_float(atomicMin((unsigned int*)addr, __float_as_uint(v)));
}

__global__ void fill_f32(float* p, float v, int n) {
  int i = blockIdx.x * blockDim.x + threadIdx.x;
  if (i < n) p[i] = v;
}

__global__ void cast_f32_bf16(const float* __restrict__ s, __bf16* __restrict__ d, int n) {
  int i = blockIdx.x * blockDim.x + threadIdx.x;
  if (i < n) d[i] = (__bf16)s[i];
}

// W [K, NC] row-major f32 -> Wt [NC, K] bf16 (column-major B for WMMA)
__global__ void transpose_cast(const float* __restrict__ W, __bf16* __restrict__ Wt,
                               int K, int NC) {
  int i = blockIdx.x * blockDim.x + threadIdx.x;
  if (i < K * NC) {
    int k = i / NC, c = i - k * NC;
    Wt[(size_t)c * K + k] = (__bf16)W[i];
  }
}

// C[M,NC] (bf16) = A[M,K](bf16,row-major) x B (Bt = [NC,K] bf16), f32 accum.
// One wave per 16x64 strip; 4 accumulators reuse the A fragment.
__global__ __launch_bounds__(128)
void wmma_gemm_bf16(const __bf16* __restrict__ A, const __bf16* __restrict__ Bt,
                    __bf16* __restrict__ C, int K, int NC) {
  const int lane = threadIdx.x & 31;
  const int wave = threadIdx.x >> 5;
  const int mt   = blockIdx.x;                       // 16-row tile
  const int n64  = blockIdx.y * (blockDim.x >> 5) + wave;
  if (n64 * 64 >= NC) return;

  const int row  = mt * 16 + (lane & 15);            // A row for this lane
  const int koff = (lane >> 4) << 4;                 // A: 0 or 16
  const int kB   = (lane >> 4) << 4;                 // B: 0 or 16
  const int colL = lane & 15;

  v8f acc[4] = {};

  const __bf16* arow = A + (size_t)row * K;
  for (int kb = 0; kb < K; kb += 32) {
    // A fragment: lanes 0-15 hold K=kb+0..15; lanes 16-31 K=kb+16..31
    union { v16bf v; v8bf p[2]; } af;
    af.p[0] = *(const v8bf*)(arow + kb + koff);
    af.p[1] = *(const v8bf*)(arow + kb + 8 + koff);
#pragma unroll
    for (int j = 0; j < 4; ++j) {
      const __bf16* bcol = Bt + (size_t)(n64 * 64 + j * 16 + colL) * K + kb + kB;
      v16bf bf = *(const v16bf*)bcol;                // 16 contiguous K values
      acc[j] = __builtin_amdgcn_wmma_f32_16x16x32_bf16(
          false, af.v, false, bf, (short)0, acc[j], false, false);
    }
  }

  const int rowD = mt * 16 + ((lane >> 4) << 3);     // C/D: lanes16-31 -> M+8
#pragma unroll
  for (int j = 0; j < 4; ++j)
#pragma unroll
    for (int v = 0; v < 8; ++v)
      C[(size_t)(rowD + v) * NC + n64 * 64 + j * 16 + colL] = (__bf16)acc[j][v];
}

// per-(node,head) wave: alpha_src/dst = <h[n,h,:], a_src/dst[h,:]>
__global__ void alpha_kernel(const __bf16* __restrict__ h,
                             const float* __restrict__ a_src,
                             const float* __restrict__ a_dst,
                             float* __restrict__ asrc, float* __restrict__ adst,
                             int Hh, int Cc) {
  int gw   = (blockIdx.x * blockDim.x + threadIdx.x) >> 5;
  int lane = threadIdx.x & 31;
  int n = gw / Hh, hd = gw - n * Hh;
  if (n >= Nn) return;
  const __bf16* hp = h + ((size_t)n * Hh + hd) * Cc;
  const float* as = a_src + hd * Cc;
  const float* ad = a_dst + hd * Cc;
  float s0 = 0.f, s1 = 0.f;
  for (int c = lane; c < Cc; c += 32) {
    float v = (float)hp[c];
    s0 += v * as[c];
    s1 += v * ad[c];
  }
  for (int off = 16; off; off >>= 1) {
    s0 += __shfl_xor(s0, off, 32);
    s1 += __shfl_xor(s1, off, 32);
  }
  if (lane == 0) { asrc[n * Hh + hd] = s0; adst[n * Hh + hd] = s1; }
}

__device__ __forceinline__ void edge_ends(const int* ei, int eid, int& s, int& d) {
  if (eid < Ee) { s = ei[eid]; d = ei[Ee + eid]; }
  else          { s = d = eid - Ee; }                // self loop
}

// e = leaky_relu(as[src]+ad[dst]); running segment max via atomics
__global__ void edge_pass1(const int* __restrict__ ei, const float* __restrict__ asrc,
                           const float* __restrict__ adst, float* __restrict__ ebuf,
                           float* __restrict__ mbuf, int Hh) {
  int t = blockIdx.x * blockDim.x + threadIdx.x;
  if (t >= ETOTe * Hh) return;
  int eid = t / Hh, hd = t - eid * Hh;
  int s, d; edge_ends(ei, eid, s, d);
  float v = asrc[s * Hh + hd] + adst[d * Hh + hd];
  v = v > 0.f ? v : 0.2f * v;
  ebuf[t] = v;
  atomicMaxFloat(&mbuf[d * Hh + hd], v);
}

// ex = exp(e - m[dst]); denom accumulate
__global__ void edge_pass2(const int* __restrict__ ei, float* __restrict__ ebuf,
                           const float* __restrict__ mbuf, float* __restrict__ den,
                           int Hh) {
  int t = blockIdx.x * blockDim.x + threadIdx.x;
  if (t >= ETOTe * Hh) return;
  int eid = t / Hh, hd = t - eid * Hh;
  int s, d; edge_ends(ei, eid, s, d); (void)s;
  float ex = __expf(ebuf[t] - mbuf[d * Hh + hd]);
  ebuf[t] = ex;
  atomicAdd(&den[d * Hh + hd], ex);
}

// y[dst,c] += (1/H) * sum_h (ex/den[dst,h]) * h[src,h,c]   (blockDim == Cc)
__global__ void edge_pass3(const int* __restrict__ ei, const float* __restrict__ ebuf,
                           const float* __restrict__ den, const __bf16* __restrict__ h,
                           float* __restrict__ y, int Hh, int Cc) {
  int eid = blockIdx.x;
  int c   = threadIdx.x;
  int s, d; edge_ends(ei, eid, s, d);
  float acc = 0.f;
  for (int hd = 0; hd < Hh; ++hd) {
    float alpha = ebuf[eid * Hh + hd] / den[d * Hh + hd];  // same-addr loads broadcast
    acc += alpha * (float)h[((size_t)s * Hh + hd) * Cc + c];
  }
  atomicAdd(&y[(size_t)d * Cc + c], acc * (1.0f / Hh));
}

// per-channel sum / sumsq  (blockDim == Cc, one thread per channel)
__global__ void bn_stats(const float* __restrict__ y, float* __restrict__ stats,
                         int n, int Cc) {
  int c = threadIdx.x;
  float s = 0.f, q = 0.f;
  for (int r = blockIdx.x; r < n; r += gridDim.x) {
    float v = y[(size_t)r * Cc + c];
    s += v; q += v * v;
  }
  atomicAdd(&stats[c], s);
  atomicAdd(&stats[Cc + c], q);
}

// BN + ELU, emit bf16 input for the next layer's WMMA GEMM
__global__ void bn_apply_elu(const float* __restrict__ y, const float* __restrict__ stats,
                             const float* __restrict__ g, const float* __restrict__ be,
                             __bf16* __restrict__ out, int n, int Cc) {
  int c = threadIdx.x;
  float mu  = stats[c] * (1.0f / n);
  float var = stats[Cc + c] * (1.0f / n) - mu * mu;
  float sc  = rsqrtf(var + 1e-5f) * g[c];
  float sh  = be[c] - mu * sc;
  for (int r = blockIdx.x; r < n; r += gridDim.x) {
    float v = y[(size_t)r * Cc + c] * sc + sh;
    v = v > 0.f ? v : (__expf(v) - 1.0f);
    out[(size_t)r * Cc + c] = (__bf16)v;
  }
}

__global__ void bias_add(float* __restrict__ out, const float* __restrict__ b,
                         int n, int Cc) {
  int i = blockIdx.x * blockDim.x + threadIdx.x;
  if (i < n * Cc) out[i] += b[i % Cc];
}

// ---------------------------------------------------------------------------

extern "C" void kernel_launch(void* const* d_in, const int* in_sizes, int n_in,
                              void* d_out, int out_size, void* d_ws, size_t ws_size,
                              hipStream_t stream) {
  (void)in_sizes; (void)n_in; (void)out_size; (void)ws_size;

  const float* x   = (const float*)d_in[0];
  const int*   ei  = (const int*)  d_in[1];
  const float* W1  = (const float*)d_in[2];
  const float* as1 = (const float*)d_in[3];
  const float* ad1 = (const float*)d_in[4];
  const float* g1  = (const float*)d_in[6];
  const float* be1 = (const float*)d_in[7];
  const float* W2  = (const float*)d_in[8];
  const float* as2 = (const float*)d_in[9];
  const float* ad2 = (const float*)d_in[10];
  const float* g2  = (const float*)d_in[12];
  const float* be2 = (const float*)d_in[13];
  const float* W3  = (const float*)d_in[14];
  const float* as3 = (const float*)d_in[15];
  const float* ad3 = (const float*)d_in[16];
  const float* b3  = (const float*)d_in[17];

  char* w = (char*)d_ws;
  __bf16* bf_in = (__bf16*)(w + OFF_BFIN);
  __bf16* bf_w  = (__bf16*)(w + OFF_BFW);
  __bf16* hbuf  = (__bf16*)(w + OFF_H);
  float*  asrc  = (float*)(w + OFF_ASRC);
  float*  adst  = (float*)(w + OFF_ADST);
  float*  mbuf  = (float*)(w + OFF_M);
  float*  den   = (float*)(w + OFF_DEN);
  float*  ebuf  = (float*)(w + OFF_E);
  float*  ybuf  = (float*)(w + OFF_Y);
  float*  stats = (float*)(w + OFF_ST);
  float*  outp  = (float*)d_out;

  auto gat = [&](const float* Wp, const float* asp, const float* adp,
                 int K, int Hh, int Cc, float* ytgt) {
    const int NC = Hh * Cc;
    transpose_cast<<<dim3((K * NC + 255) / 256), 256, 0, stream>>>(Wp, bf_w, K, NC);
    dim3 ggrid(Nn / 16, (NC / 64 + 3) / 4);
    wmma_gemm_bf16<<<ggrid, 128, 0, stream>>>(bf_in, bf_w, hbuf, K, NC);
    alpha_kernel<<<dim3((Nn * Hh * 32 + 255) / 256), 256, 0, stream>>>(
        hbuf, asp, adp, asrc, adst, Hh, Cc);
    fill_f32<<<dim3((Nn * Hh + 255) / 256), 256, 0, stream>>>(mbuf, -INFINITY, Nn * Hh);
    fill_f32<<<dim3((Nn * Hh + 255) / 256), 256, 0, stream>>>(den, 0.f, Nn * Hh);
    fill_f32<<<dim3((Nn * Cc + 255) / 256), 256, 0, stream>>>(ytgt, 0.f, Nn * Cc);
    edge_pass1<<<dim3((ETOTe * Hh + 255) / 256), 256, 0, stream>>>(ei, asrc, adst, ebuf, mbuf, Hh);
    edge_pass2<<<dim3((ETOTe * Hh + 255) / 256), 256, 0, stream>>>(ei, ebuf, mbuf, den, Hh);
    edge_pass3<<<dim3(ETOTe), Cc, 0, stream>>>(ei, ebuf, den, hbuf, ytgt, Hh, Cc);
  };

  // ---- Layer 1: GATConv(64 -> 128, H=4, mean) + BN + ELU  (b1 cancels in BN)
  cast_f32_bf16<<<dim3((Nn * 64 + 255) / 256), 256, 0, stream>>>(x, bf_in, Nn * 64);
  gat(W1, as1, ad1, 64, 4, 128, ybuf);
  fill_f32<<<1, 256, 0, stream>>>(stats, 0.f, 256);
  bn_stats<<<dim3(256), 128, 0, stream>>>(ybuf, stats, Nn, 128);
  bn_apply_elu<<<dim3(256), 128, 0, stream>>>(ybuf, stats, g1, be1, bf_in, Nn, 128);

  // ---- Layer 2: GATConv(128 -> 128, H=4, mean) + BN + ELU (b2 cancels in BN)
  gat(W2, as2, ad2, 128, 4, 128, ybuf);
  fill_f32<<<1, 256, 0, stream>>>(stats, 0.f, 256);
  bn_stats<<<dim3(256), 128, 0, stream>>>(ybuf, stats, Nn, 128);
  bn_apply_elu<<<dim3(256), 128, 0, stream>>>(ybuf, stats, g2, be2, bf_in, Nn, 128);

  // ---- Layer 3: GATConv(128 -> 64, H=1) + b3, straight into d_out
  gat(W3, as3, ad3, 128, 1, 64, outp);
  bias_add<<<dim3((Nn * 64 + 255) / 256), 256, 0, stream>>>(outp, b3, Nn, 64);
}
